// Conv2DUF_1675037245808
// MI455X (gfx1250) — matrix operand order repaired and checked by reading, not compiled, not run on hardware
//
#include <hip/hip_runtime.h>

// ---------------------------------------------------------------------------
// Implicit-GEMM 3x3 conv (im2col fused) for gfx1250, split-bf16 WMMA (bf16x3).
//   M = B*H*W = 100352, N = 256, K = 1152 (iterated as 9 kpos x 4 c-chunks)
//   Pass 1: x  f32 NCHW -> zero-padded NHWC split bf16  xh/xl[b][58][58][c]
//           w  f32 [k][o] -> split bf16, transposed wT[o][kpos][c]
//   Pass 2: GEMM. A and B tiles are PURE async copies global->LDS
//           (global_load_async_to_lds_b64, ASYNCcnt): no masking, no
//           conversions, no ds stores in the hot loop.
//   3 WMMA passes ordered (Ahi*Bhi), (Alo*Bhi), (Ahi*Blo) -> ~fp32 accuracy.
// ---------------------------------------------------------------------------

typedef __attribute__((ext_vector_type(16))) __bf16 v16bf;
typedef __attribute__((ext_vector_type(8)))  float  v8f;

#define C_IN   128
#define H_IMG  56
#define W_IMG  56
#define L_IMG  (H_IMG * W_IMG)      // 3136 pixels per image
#define HP     58                   // padded height/width
#define O_CH   256
#define N_BATCH 32
#define XP_ELEMS (N_BATCH * HP * HP * C_IN)  // 13,778,944 per (hi|lo) array
#define W_ELEMS  (1152 * O_CH)               // 294,912
#define M_TILE 128
#define N_TILE 128
#define KC     32                   // K per WMMA (bf16)
#define LDSTR  40                   // padded LDS row stride (elems) -> 80B, 16B aligned

__device__ __forceinline__ unsigned short f2bf_rn(float f) {
  unsigned u = __builtin_bit_cast(unsigned, f);
  u += 0x7FFFu + ((u >> 16) & 1u);  // round-to-nearest-even
  return (unsigned short)(u >> 16);
}
__device__ __forceinline__ float bf2f(unsigned short h) {
  unsigned u = ((unsigned)h) << 16;
  return __builtin_bit_cast(float, u);
}

// Async global->LDS copy, 8 bytes per lane (CDNA5; tracked on ASYNCcnt).
__device__ __forceinline__ void async_copy_b64(unsigned lds_off, const void* g) {
  asm volatile("global_load_async_to_lds_b64 %0, %1, off"
               :: "v"(lds_off), "v"(g) : "memory");
}
__device__ __forceinline__ void wait_asynccnt0() {
  asm volatile("s_wait_asynccnt 0x0" ::: "memory");
}

// ---- pass 1a: x NCHW f32 -> padded NHWC split bf16 (LDS-tiled transpose) ---
// grid: (3136/32 pixel-tiles, 128/32 channel-tiles, 32 batches), 256 threads.
__global__ __launch_bounds__(256)
void pack_x_nhwc_pad(const float* __restrict__ x,
                     unsigned short* __restrict__ xh,
                     unsigned short* __restrict__ xl) {
  __shared__ float tile[32][33];
  const int tid = threadIdx.x;
  const int l0  = blockIdx.x * 32;
  const int c0  = blockIdx.y * 32;
  const int b   = blockIdx.z;

  // read: lanes sweep pixels (coalesced per channel row)
#pragma unroll
  for (int rr = 0; rr < 4; ++rr) {
    const int cp = rr * 8 + (tid >> 5);
    const int pp = tid & 31;
    tile[cp][pp] = x[((size_t)b * C_IN + (c0 + cp)) * L_IMG + l0 + pp];
  }
  __syncthreads();
  // write: lanes sweep channels (contiguous 64B per pixel in NHWC)
#pragma unroll
  for (int rr = 0; rr < 4; ++rr) {
    const int pp = rr * 8 + (tid >> 5);
    const int cp = tid & 31;
    const float v = tile[cp][pp];
    const int l  = l0 + pp;
    const int ho = l / W_IMG;
    const int wo = l - ho * W_IMG;
    const size_t dst =
        (((size_t)b * HP + (ho + 1)) * HP + (wo + 1)) * C_IN + c0 + cp;
    const unsigned short hi = f2bf_rn(v);
    xh[dst] = hi;
    xl[dst] = f2bf_rn(v - bf2f(hi));
  }
}

// ---- pass 1b: w [k= c*9+kpos][o] f32 -> split bf16 wT[o][kpos][c] ----------
__global__ __launch_bounds__(256)
void pack_w_split_t(const float* __restrict__ wt,
                    unsigned short* __restrict__ whi,
                    unsigned short* __restrict__ wlo) {
  const int i = blockIdx.x * 256 + threadIdx.x;   // i = k*256 + o
  const int k = i >> 8;
  const int o = i & 255;
  const int c = k / 9;
  const int kpos = k - c * 9;
  const int dst = (o * 9 + kpos) * C_IN + c;
  const float f = wt[i];
  const unsigned short hi = f2bf_rn(f);
  whi[dst] = hi;
  wlo[dst] = f2bf_rn(f - bf2f(hi));
}

// A fragment (16x32 bf16): lanes 0-15 -> K[0..7],[16..23]; lanes 16-31 -> K[8..15],[24..31]
__device__ __forceinline__ v16bf load_frag_a(const unsigned short* row, int half) {
  union { uint4 q[2]; v16bf v; } u;
  const unsigned short* p = row + half * 8;
  u.q[0] = *(const uint4*)(p);
  u.q[1] = *(const uint4*)(p + 16);
  return u.v;
}
// B fragment (32x16 bf16, stored transposed [n][k]): lanes 0-15 -> K[0..7],[16..23]; 16-31 -> K[8..15],[24..31]
__device__ __forceinline__ v16bf load_frag_b(const unsigned short* row, int half) {
  union { uint4 q[2]; v16bf v; } u;
  const unsigned short* p = row + half * 8;
  u.q[0] = *(const uint4*)(p);
  u.q[1] = *(const uint4*)(p + 16);
  return u.v;
}
__device__ __forceinline__ v8f vzero8() {
  v8f z = {0.f, 0.f, 0.f, 0.f, 0.f, 0.f, 0.f, 0.f};
  return z;
}

// ---- pass 2: implicit GEMM ----
template <bool PACKED>
__global__ __launch_bounds__(256)
void conv3x3_igemm(const void* __restrict__ xhin,  // PACKED: bf16 NHWC-pad hi ; else f32 NCHW x
                   const void* __restrict__ xlin,  // PACKED: bf16 NHWC-pad lo
                   const void* __restrict__ whiin, // PACKED: bf16 wT[o][kpos][c]; else f32 w[k][o]
                   const void* __restrict__ wloin, // PACKED only
                   float* __restrict__ out) {
  __shared__ __align__(16) unsigned short AsHi[M_TILE * LDSTR];
  __shared__ __align__(16) unsigned short AsLo[M_TILE * LDSTR];
  __shared__ __align__(16) unsigned short BsHi[N_TILE * LDSTR];
  __shared__ __align__(16) unsigned short BsLo[N_TILE * LDSTR];

  const unsigned short* xh  = (const unsigned short*)xhin;
  const unsigned short* xl  = (const unsigned short*)xlin;
  const float*          xf  = (const float*)xhin;
  const unsigned short* whi = (const unsigned short*)whiin;
  const unsigned short* wlo = (const unsigned short*)wloin;
  const float*          wf  = (const float*)whiin;

  const int tid    = threadIdx.x;
  const int lane   = tid & 31;
  const int wave   = tid >> 5;        // 0..7
  const int wm     = wave & 3;        // M wave group (32 rows each)
  const int wn     = wave >> 2;       // N wave group (64 cols each)
  const int lane16 = lane & 15;
  const int half   = lane >> 4;

  const int m0 = blockIdx.x * M_TILE; // global pixel base
  const int o0 = blockIdx.y * N_TILE; // output-channel base

  // ---- PACKED: async A-copy geometry: 128 rows x 64B; 4 x 8B per thread ----
  int      aBaseEl[4];   // (padded-pixel)*128 + sc*4 (add shift*128 + c0 per chunk)
  unsigned aLds[4];      // LDS byte offset within tile
  const unsigned ahiBase = (unsigned)(size_t)(&AsHi[0]);
  const unsigned aloBase = (unsigned)(size_t)(&AsLo[0]);
  const unsigned bhiBase = (unsigned)(size_t)(&BsHi[0]);
  const unsigned bloBase = (unsigned)(size_t)(&BsLo[0]);
#pragma unroll
  for (int r = 0; r < 4; ++r) {
    const int seg = r * 256 + tid;    // 0..1023
    const int mA  = seg >> 3;
    const int sc  = seg & 7;
    const int gm2 = m0 + mA;
    const int b2  = gm2 / L_IMG;
    const int l2  = gm2 - b2 * L_IMG;
    const int ho2 = l2 / W_IMG;
    const int wo2 = l2 - ho2 * W_IMG;
    aBaseEl[r] = (((b2 * HP) + ho2 + 1) * HP + wo2 + 1) * C_IN + sc * 4;
    aLds[r]    = (unsigned)(mA * LDSTR + sc * 4) * 2u;
  }

  // ---- fallback A geometry (masked im2col from f32 NCHW) ----
  const int mm  = tid & 127;
  const int tHi = tid >> 7;
  const int gm  = m0 + mm;
  const int bA  = gm / L_IMG;
  const int lA  = gm - bA * L_IMG;
  const int hoA = lA / W_IMG;
  const int woA = lA - hoA * W_IMG;
  const size_t xBatch = (size_t)bA * C_IN * L_IMG;

  v8f acc[2][4];
#pragma unroll
  for (int mi = 0; mi < 2; ++mi)
#pragma unroll
    for (int ni = 0; ni < 4; ++ni) acc[mi][ni] = vzero8();

  for (int kpos = 0; kpos < 9; ++kpos) {
    const int dh = kpos / 3 - 1;
    const int dw = kpos % 3 - 1;
    const int shiftEl = (dh * HP + dw) * C_IN;  // uniform padded-NHWC shift
    const int hA = hoA + dh;
    const int wA = woA + dw;
    const bool okA = (hA >= 0) && (hA < H_IMG) && (wA >= 0) && (wA < W_IMG);
    const size_t baseA = okA ? (xBatch + (size_t)(hA * W_IMG + wA)) : (size_t)0;

    for (int cc = 0; cc < 4; ++cc) {
      const int c0 = cc * KC;

      if (PACKED) {
        // ---- A tile: pure async copy (pre-masked by zero padding) ----
#pragma unroll
        for (int r = 0; r < 4; ++r) {
          const size_t g = (size_t)(aBaseEl[r] + shiftEl + c0);
          async_copy_b64(ahiBase + aLds[r], xh + g);
          async_copy_b64(aloBase + aLds[r], xl + g);
        }
        // ---- B tile: pure async copy ----
#pragma unroll
        for (int r = 0; r < 4; ++r) {
          const int seg = r * 256 + tid;
          const int n   = seg >> 3;
          const int sc  = seg & 7;
          const size_t goff = ((size_t)(o0 + n) * 9 + kpos) * C_IN + c0 + sc * 4;
          const unsigned loff = (unsigned)(n * LDSTR + sc * 4) * 2u;
          async_copy_b64(bhiBase + loff, whi + goff);
          async_copy_b64(bloBase + loff, wlo + goff);
        }
        wait_asynccnt0();
      } else {
        // ---- fallback fills with in-kernel f32 split ----
#pragma unroll
        for (int r = 0; r < 16; ++r) {
          const int cp = r * 2 + tHi;
          float v = xf[baseA + (size_t)(c0 + cp) * L_IMG];
          v = okA ? v : 0.f;
          const unsigned short hi = f2bf_rn(v);
          AsHi[mm * LDSTR + cp] = hi;
          AsLo[mm * LDSTR + cp] = f2bf_rn(v - bf2f(hi));
        }
#pragma unroll
        for (int r = 0; r < 16; ++r) {
          const int cp = r * 2 + tHi;
          const float v = wf[(size_t)((c0 + cp) * 9 + kpos) * O_CH + (o0 + mm)];
          const unsigned short hi = f2bf_rn(v);
          BsHi[mm * LDSTR + cp] = hi;
          BsLo[mm * LDSTR + cp] = f2bf_rn(v - bf2f(hi));
        }
      }
      __syncthreads();

      // ---- frags: A loaded once; B set swapped once (pass order
      //      Ahi*Bhi, Alo*Bhi, Ahi*Blo) ----
      v16bf ahi[2], alo[2], bfr[4];
#pragma unroll
      for (int mi = 0; mi < 2; ++mi) {
        const int row = wm * 32 + mi * 16 + lane16;
        ahi[mi] = load_frag_a(AsHi + row * LDSTR, half);
        alo[mi] = load_frag_a(AsLo + row * LDSTR, half);
      }
#pragma unroll
      for (int ni = 0; ni < 4; ++ni) {
        const int row = wn * 64 + ni * 16 + lane16;
        bfr[ni] = load_frag_b(BsHi + row * LDSTR, half);
      }
#pragma unroll
      for (int mi = 0; mi < 2; ++mi)
#pragma unroll
        for (int ni = 0; ni < 4; ++ni)
          acc[mi][ni] = __builtin_amdgcn_wmma_f32_16x16x32_bf16(
              false, ahi[mi], false, bfr[ni], (short)0, acc[mi][ni], false, false);
#pragma unroll
      for (int mi = 0; mi < 2; ++mi)
#pragma unroll
        for (int ni = 0; ni < 4; ++ni)
          acc[mi][ni] = __builtin_amdgcn_wmma_f32_16x16x32_bf16(
              false, alo[mi], false, bfr[ni], (short)0, acc[mi][ni], false, false);
#pragma unroll
      for (int ni = 0; ni < 4; ++ni) {
        const int row = wn * 64 + ni * 16 + lane16;
        bfr[ni] = load_frag_b(BsLo + row * LDSTR, half);
      }
#pragma unroll
      for (int mi = 0; mi < 2; ++mi)
#pragma unroll
        for (int ni = 0; ni < 4; ++ni)
          acc[mi][ni] = __builtin_amdgcn_wmma_f32_16x16x32_bf16(
              false, ahi[mi], false, bfr[ni], (short)0, acc[mi][ni], false, false);

      __syncthreads();
    }
  }

  // ---- store: lane = 16*(m>=8)+n, VGPR r = m%8 -> 8 consecutive pixels.
  // Runs start at multiples of 8 and 3136 % 8 == 0 -> never straddle an
  // image boundary: two aligned B128 stores per 16x16 tile.
#pragma unroll
  for (int mi = 0; mi < 2; ++mi) {
    const int mbase = m0 + wm * 32 + mi * 16 + half * 8;  // multiple of 8
    const int bb = mbase / L_IMG;
    const int lb = mbase - bb * L_IMG;                    // multiple of 8
#pragma unroll
    for (int ni = 0; ni < 4; ++ni) {
      const int og = o0 + wn * 64 + ni * 16 + lane16;
      float* p = out + ((size_t)bb * O_CH + og) * L_IMG + lb;
      const v8f a = acc[mi][ni];
      *(float4*)(p)     = make_float4(a[0], a[1], a[2], a[3]);
      *(float4*)(p + 4) = make_float4(a[4], a[5], a[6], a[7]);
    }
  }
}

extern "C" void kernel_launch(void* const* d_in, const int* in_sizes, int n_in,
                              void* d_out, int out_size, void* d_ws, size_t ws_size,
                              hipStream_t stream) {
  const float* x  = (const float*)d_in[0];   // [32,128,56,56] f32
  const float* wt = (const float*)d_in[1];   // [1152,256] f32
  float* out      = (float*)d_out;           // [32,256,56,56] f32

  dim3 grid(100352 / M_TILE, O_CH / N_TILE, 1); // (784, 2)
  dim3 block(256, 1, 1);

  // ws layout: [xh bf16][xl bf16][whiT bf16][wloT bf16]
  const size_t xpadBytes = (size_t)XP_ELEMS * 2;
  const size_t need = 2 * xpadBytes + (size_t)W_ELEMS * 2 * 2;
  if (ws_size >= need) {
    unsigned short* xh   = (unsigned short*)d_ws;
    unsigned short* xlo  = xh + XP_ELEMS;
    unsigned short* whiT = xlo + XP_ELEMS;
    unsigned short* wloT = whiT + W_ELEMS;
    hipMemsetAsync(d_ws, 0, 2 * xpadBytes, stream);  // zero padded borders
    pack_x_nhwc_pad<<<dim3(L_IMG / 32, C_IN / 32, N_BATCH), block, 0, stream>>>(x, xh, xlo);
    pack_w_split_t<<<W_ELEMS / 256, block, 0, stream>>>(wt, whiT, wloT);
    conv3x3_igemm<true><<<grid, block, 0, stream>>>(xh, xlo, whiT, wloT, out);
  } else {
    conv3x3_igemm<false><<<grid, block, 0, stream>>>(x, nullptr, wt, nullptr, out);
  }
}
